// RGCN_14826227106512
// MI455X (gfx1250) — matrix operand order repaired and checked by reading, not compiled, not run on hardware
//
#include <hip/hip_runtime.h>

#define D 128

typedef __attribute__((ext_vector_type(16))) __bf16 v16bf;
typedef __attribute__((ext_vector_type(8)))  float  v8f;

union FragAB {
    uint4 q[2];
    v16bf v;
};

__device__ __forceinline__ unsigned short f2bf(float f) {
    unsigned int u = __float_as_uint(f);
    u += 0x7FFFu + ((u >> 16) & 1u);      // round-to-nearest-even to bf16
    return (unsigned short)(u >> 16);
}

// ---------------- degree kernels ----------------
__global__ void k_count_deg(const int* __restrict__ idx, float* __restrict__ deg, int nE) {
    int e = blockIdx.x * blockDim.x + threadIdx.x;
    if (e < nE) atomicAdd(&deg[idx[e]], 1.0f);
}

__global__ void k_rsqrt_deg(float* __restrict__ deg, int n) {
    int i = blockIdx.x * blockDim.x + threadIdx.x;
    if (i < n) deg[i] = rsqrtf(fmaxf(deg[i], 1.0f));
}

// ---- pack W[128,128] f32 row-major into WMMA B-fragment order (bf16 pair / dword) ----
// dword index i = ((ct*4 + ks)*32 + lane)*8 + v ; lane<16: N=lane,K=ks*32+2v ; lane>=16: K += 16
__global__ void k_pack_weight(const float* __restrict__ W, unsigned int* __restrict__ Bp) {
    int i = blockIdx.x * blockDim.x + threadIdx.x;   // 8192 dwords total
    if (i >= 8192) return;
    int v    = i & 7;
    int lane = (i >> 3) & 31;
    int ks   = (i >> 8) & 3;
    int ct   = i >> 10;
    int n    = lane & 15;
    int kh   = lane >> 4;
    int k    = ks * 32 + kh * 16 + 2 * v;
    int col  = ct * 16 + n;
    unsigned int lo = f2bf(W[k * D + col]);
    unsigned int hi = f2bf(W[(k + 1) * D + col]);
    Bp[i] = lo | (hi << 16);
}

// ---- out-degree scale + f32 -> bf16 convert (rs == nullptr -> scale 1) ----
__global__ void k_scale_bf16(const float* __restrict__ X, const float* __restrict__ rs,
                             unsigned short* __restrict__ Xb, int rows) {
    int i = blockIdx.x * blockDim.x + threadIdx.x;   // rows*32 threads, 4 elems each
    if (i >= rows * 32) return;
    int row = i >> 5;
    int c   = (i & 31) * 4;
    float s = rs ? rs[row] : 1.0f;
    float4 x = *(const float4*)(X + (long long)row * D + c);
    uint2 o;
    o.x = (unsigned int)f2bf(x.x * s) | ((unsigned int)f2bf(x.y * s) << 16);
    o.y = (unsigned int)f2bf(x.z * s) | ((unsigned int)f2bf(x.w * s) << 16);
    *(uint2*)(Xb + (long long)row * D + c) = o;
}

// ---- H[M,128] = Xb[M,128](bf16) @ Wpacked(bf16), f32 accumulate, WMMA 16x16x32 ----
__global__ void __launch_bounds__(128)
k_gemm_bf16(const unsigned short* __restrict__ Xb, const unsigned int* __restrict__ Bp,
            float* __restrict__ C, int M) {
    const int wave = threadIdx.x >> 5;
    const int lane = threadIdx.x & 31;
    const int row0 = blockIdx.x * 64 + wave * 16;
    if (row0 >= M) return;                       // wave-uniform; EXEC stays all-1s
    const int mloc  = lane & 15;
    const int khalf = lane >> 4;
    const int arow  = min(row0 + mloc, M - 1);   // clamp tail rows (stores are guarded)
    const unsigned short* arowp = Xb + (long long)arow * D;

    v8f acc[8] = {};
    #pragma unroll
    for (int ks = 0; ks < 4; ++ks) {
        const int kk = ks * 32;
        FragAB a;
        a.q[0] = *(const uint4*)(arowp + kk + khalf * 16);       // K = kk+kh*16 .. +7
        a.q[1] = *(const uint4*)(arowp + kk + khalf * 16 + 8);   // K = kk+kh*16+8 .. +15
        #pragma unroll
        for (int ct = 0; ct < 8; ++ct) {
            FragAB b;
            const uint4* bp = (const uint4*)(Bp + (((ct * 4 + ks) * 32 + lane) * 8));
            b.q[0] = bp[0];
            b.q[1] = bp[1];
            acc[ct] = __builtin_amdgcn_wmma_f32_16x16x32_bf16(
                false, a.v, false, b.v, (short)0, acc[ct], false, false);
        }
    }
    #pragma unroll
    for (int ct = 0; ct < 8; ++ct) {
        const int col = ct * 16 + mloc;
        #pragma unroll
        for (int r = 0; r < 8; ++r) {
            int rr = row0 + khalf * 8 + r;       // D layout: VGPR r -> M=r (+8 for hi half)
            if (rr < M) C[(long long)rr * D + col] = acc[ct][r];
        }
    }
}

// ---- edge scatter: Agg[dst] += H[src], one wave per edge, float4 per lane ----
__global__ void k_scatter(const float* __restrict__ H, const int* __restrict__ src,
                          const int* __restrict__ dst, float* __restrict__ Agg, int nE) {
    long long t = (long long)blockIdx.x * blockDim.x + threadIdx.x;
    long long e = t >> 5;
    int lane = (int)(t & 31);
    if (e >= nE) return;
    int s = src[e], d = dst[e];
    float4 h = *(const float4*)(H + (long long)s * D + lane * 4);
    float* p = Agg + (long long)d * D + lane * 4;
    atomicAdd(p + 0, h.x);
    atomicAdd(p + 1, h.y);
    atomicAdd(p + 2, h.z);
    atomicAdd(p + 3, h.w);
}

// ---- out = [relu]( Agg * rs_in + bias ) [+= existing out] ----
__global__ void k_finalize(const float* __restrict__ Agg, const float* __restrict__ rs,
                           const float* __restrict__ bias, float* __restrict__ outp,
                           int rows, int relu, int accum) {
    int i = blockIdx.x * blockDim.x + threadIdx.x;   // rows*32 threads
    if (i >= rows * 32) return;
    int row = i >> 5;
    int c   = (i & 31) * 4;
    float s = rs ? rs[row] : 1.0f;
    float4 a = *(const float4*)(Agg + (long long)row * D + c);
    float4 b = *(const float4*)(bias + c);
    float4 r;
    r.x = a.x * s + b.x;  r.y = a.y * s + b.y;
    r.z = a.z * s + b.z;  r.w = a.w * s + b.w;
    if (relu) {
        r.x = fmaxf(r.x, 0.0f); r.y = fmaxf(r.y, 0.0f);
        r.z = fmaxf(r.z, 0.0f); r.w = fmaxf(r.w, 0.0f);
    }
    float* op = outp + (long long)row * D + c;
    if (accum) {
        float4 p = *(const float4*)op;
        r.x += p.x; r.y += p.y; r.z += p.z; r.w += p.w;
    }
    *(float4*)op = r;
}

extern "C" void kernel_launch(void* const* d_in, const int* in_sizes, int n_in,
                              void* d_out, int out_size, void* d_ws, size_t ws_size,
                              hipStream_t stream) {
    (void)n_in; (void)out_size; (void)ws_size;
    const int ng = in_sizes[0] / D, nc = in_sizes[1] / D, nt = in_sizes[2] / D;
    const int Egc = in_sizes[31], Egt = in_sizes[35];

    const float* x_cell   = (const float*)d_in[1];
    const float* x_gotem  = (const float*)d_in[2];
    const float* W_eg_c2g = (const float*)d_in[3];  const float* b_eg_c2g = (const float*)d_in[4];
    const float* W_eg_t2g = (const float*)d_in[5];  const float* b_eg_t2g = (const float*)d_in[6];
    const float* W1_g2c   = (const float*)d_in[11]; const float* b1_g2c   = (const float*)d_in[12];
    const float* W1_g2t   = (const float*)d_in[15]; const float* b1_g2t   = (const float*)d_in[16];
    const float* W2_c2g   = (const float*)d_in[21]; const float* b2_c2g   = (const float*)d_in[22];
    const float* W2_t2g   = (const float*)d_in[25]; const float* b2_t2g   = (const float*)d_in[26];
    const float* W3_g2c   = (const float*)d_in[27]; const float* b3_g2c   = (const float*)d_in[28];
    const float* Wd       = (const float*)d_in[29]; const float* bd       = (const float*)d_in[30];
    const int* eg2c_src = (const int*)d_in[31]; const int* eg2c_dst = (const int*)d_in[32];
    const int* ec2g_src = (const int*)d_in[33]; const int* ec2g_dst = (const int*)d_in[34];
    const int* eg2t_src = (const int*)d_in[35]; const int* eg2t_dst = (const int*)d_in[36];
    const int* et2g_src = (const int*)d_in[37]; const int* et2g_dst = (const int*)d_in[38];

    char* ws = (char*)d_ws;
    size_t off = 0;
    auto alloc = [&](size_t bytes) -> void* {
        void* p = ws + off;
        off = (off + bytes + 255) & ~(size_t)255;
        return p;
    };

    // rsqrt-degree arrays (contiguous -> single memset)
    size_t degOff = off;
    float* rs_ec2g_src = (float*)alloc((size_t)nc * 4);
    float* rs_ec2g_dst = (float*)alloc((size_t)ng * 4);
    float* rs_et2g_src = (float*)alloc((size_t)nt * 4);
    float* rs_et2g_dst = (float*)alloc((size_t)ng * 4);
    float* rs_eg2c_src = (float*)alloc((size_t)ng * 4);
    float* rs_eg2c_dst = (float*)alloc((size_t)nc * 4);
    float* rs_eg2t_src = (float*)alloc((size_t)ng * 4);
    float* rs_eg2t_dst = (float*)alloc((size_t)nt * 4);
    size_t degBytes = off - degOff;

    // packed bf16 weights (8192 dwords each)
    unsigned int* Bp_eg_c2g = (unsigned int*)alloc(8192 * 4);
    unsigned int* Bp_eg_t2g = (unsigned int*)alloc(8192 * 4);
    unsigned int* Bp_1_g2c  = (unsigned int*)alloc(8192 * 4);
    unsigned int* Bp_1_g2t  = (unsigned int*)alloc(8192 * 4);
    unsigned int* Bp_2_c2g  = (unsigned int*)alloc(8192 * 4);
    unsigned int* Bp_2_t2g  = (unsigned int*)alloc(8192 * 4);
    unsigned int* Bp_3_g2c  = (unsigned int*)alloc(8192 * 4);
    unsigned int* Bp_d      = (unsigned int*)alloc(8192 * 4);

    unsigned short* Xbf = (unsigned short*)alloc((size_t)nc * D * 2);  // max n_src
    float* H    = (float*)alloc((size_t)nc * D * 4);                   // GEMM out (max)
    float* AggG = (float*)alloc((size_t)ng * D * 4);
    float* AggC = (float*)alloc((size_t)nc * D * 4);
    float* AggT = (float*)alloc((size_t)nt * D * 4);
    float* g    = (float*)alloc((size_t)ng * D * 4);
    float* c1   = (float*)alloc((size_t)nc * D * 4);
    float* t1   = (float*)alloc((size_t)nt * D * 4);
    float* g2   = (float*)alloc((size_t)ng * D * 4);
    float* c3   = (float*)alloc((size_t)nc * D * 4);

    // ---- degrees (edge-structure only, once) ----
    hipMemsetAsync(ws + degOff, 0, degBytes, stream);
    k_count_deg<<<(Egc + 255) / 256, 256, 0, stream>>>(ec2g_src, rs_ec2g_src, Egc);
    k_count_deg<<<(Egc + 255) / 256, 256, 0, stream>>>(ec2g_dst, rs_ec2g_dst, Egc);
    k_count_deg<<<(Egt + 255) / 256, 256, 0, stream>>>(et2g_src, rs_et2g_src, Egt);
    k_count_deg<<<(Egt + 255) / 256, 256, 0, stream>>>(et2g_dst, rs_et2g_dst, Egt);
    k_count_deg<<<(Egc + 255) / 256, 256, 0, stream>>>(eg2c_src, rs_eg2c_src, Egc);
    k_count_deg<<<(Egc + 255) / 256, 256, 0, stream>>>(eg2c_dst, rs_eg2c_dst, Egc);
    k_count_deg<<<(Egt + 255) / 256, 256, 0, stream>>>(eg2t_src, rs_eg2t_src, Egt);
    k_count_deg<<<(Egt + 255) / 256, 256, 0, stream>>>(eg2t_dst, rs_eg2t_dst, Egt);
    int nDeg = (int)(degBytes / 4);
    k_rsqrt_deg<<<(nDeg + 255) / 256, 256, 0, stream>>>((float*)(ws + degOff), nDeg);

    // ---- pack weights (once) ----
    k_pack_weight<<<32, 256, 0, stream>>>(W_eg_c2g, Bp_eg_c2g);
    k_pack_weight<<<32, 256, 0, stream>>>(W_eg_t2g, Bp_eg_t2g);
    k_pack_weight<<<32, 256, 0, stream>>>(W1_g2c,   Bp_1_g2c);
    k_pack_weight<<<32, 256, 0, stream>>>(W1_g2t,   Bp_1_g2t);
    k_pack_weight<<<32, 256, 0, stream>>>(W2_c2g,   Bp_2_c2g);
    k_pack_weight<<<32, 256, 0, stream>>>(W2_t2g,   Bp_2_t2g);
    k_pack_weight<<<32, 256, 0, stream>>>(W3_g2c,   Bp_3_g2c);
    k_pack_weight<<<32, 256, 0, stream>>>(Wd,       Bp_d);

    auto gconv = [&](const float* X, const float* rs_out, const unsigned int* Bpw,
                     const int* src, const int* dst, const float* rs_in, const float* bias,
                     float* Agg, float* outp, int n_src, int n_dst, int nE, int relu, int accum) {
        k_scale_bf16<<<(n_src * 32 + 255) / 256, 256, 0, stream>>>(X, rs_out, Xbf, n_src);
        k_gemm_bf16<<<(n_src + 63) / 64, 128, 0, stream>>>(Xbf, Bpw, H, n_src);
        hipMemsetAsync(Agg, 0, (size_t)n_dst * D * 4, stream);
        long long tot = (long long)nE * 32;
        k_scatter<<<(unsigned)((tot + 255) / 256), 256, 0, stream>>>(H, src, dst, Agg, nE);
        k_finalize<<<(n_dst * 32 + 255) / 256, 256, 0, stream>>>(Agg, rs_in, bias, outp, n_dst, relu, accum);
    };

    // Live dataflow only (h0_cell/c/t/g1/c2 are dead w.r.t. the returned value):
    // g  = relu( gconv(x_cell, ec2g) + gconv(x_gotem, et2g) )   (outer relu idempotent)
    gconv(x_cell,  rs_ec2g_src, Bp_eg_c2g, ec2g_src, ec2g_dst, rs_ec2g_dst, b_eg_c2g, AggG, g,  nc, ng, Egc, 1, 0);
    gconv(x_gotem, rs_et2g_src, Bp_eg_t2g, et2g_src, et2g_dst, rs_et2g_dst, b_eg_t2g, AggG, g,  nt, ng, Egt, 1, 1);
    // c1 = gconv(g, eg2c) ; t1 = gconv(g, eg2t)
    gconv(g,  rs_eg2c_src, Bp_1_g2c, eg2c_src, eg2c_dst, rs_eg2c_dst, b1_g2c, AggC, c1, ng, nc, Egc, 1, 0);
    gconv(g,  rs_eg2t_src, Bp_1_g2t, eg2t_src, eg2t_dst, rs_eg2t_dst, b1_g2t, AggT, t1, ng, nt, Egt, 1, 0);
    // g2 = gconv(c1, ec2g) + gconv(t1, et2g)
    gconv(c1, rs_ec2g_src, Bp_2_c2g, ec2g_src, ec2g_dst, rs_ec2g_dst, b2_c2g, AggG, g2, nc, ng, Egc, 1, 0);
    gconv(t1, rs_et2g_src, Bp_2_t2g, et2g_src, et2g_dst, rs_et2g_dst, b2_t2g, AggG, g2, nt, ng, Egt, 1, 1);
    // c3 = gconv(g2, eg2c)
    gconv(g2, rs_eg2c_src, Bp_3_g2c, eg2c_src, eg2c_dst, rs_eg2c_dst, b3_g2c, AggC, c3, ng, nc, Egc, 1, 0);

    // out = c3 @ Wd + bd (no degree scale, no relu)
    k_scale_bf16<<<(nc * 32 + 255) / 256, 256, 0, stream>>>(c3, nullptr, Xbf, nc);
    k_gemm_bf16<<<(nc + 63) / 64, 128, 0, stream>>>(Xbf, Bp_d, H, nc);
    k_finalize<<<(nc * 32 + 255) / 256, 256, 0, stream>>>(H, nullptr, bd, (float*)d_out, nc, 0, 0);
}
